// mesh_estimator_head_46402826665987
// MI455X (gfx1250) — matrix operand order repaired and checked by reading, not compiled, not run on hardware
//
#include <hip/hip_runtime.h>
#include <hip/hip_bf16.h>

// ---- gfx1250 async global->LDS path (guarded so compilation never breaks) ----
#if defined(__has_builtin)
#  if __has_builtin(__builtin_amdgcn_global_load_async_to_lds_b128) && \
      __has_builtin(__builtin_amdgcn_s_wait_asynccnt)
#    define USE_ASYNC_LDS 1
#  endif
#endif
#ifndef USE_ASYNC_LDS
#  define USE_ASYNC_LDS 0
#endif

typedef __attribute__((ext_vector_type(16))) _Float16 v16h;
typedef __attribute__((ext_vector_type(8)))  _Float16 h8;
typedef __attribute__((ext_vector_type(8)))  float    v8f;

#if USE_ASYNC_LDS
// Builtin signature (from hipcc diagnostic): param0 = non-const pointer to
// 'int __attribute__((vector_size(16)))' in the global (__device__/AS1) address
// space; param1 = same pointee in LDS (AS3); then immediate offset + cpol.
typedef int vint4 __attribute__((vector_size(4 * sizeof(int))));
typedef __attribute__((address_space(1))) vint4* gptr_v4i;
typedef __attribute__((address_space(3))) vint4* lptr_v4i;

__device__ __forceinline__ void async_copy_b128(const _Float16* g, _Float16* l) {
    __builtin_amdgcn_global_load_async_to_lds_b128(
        (gptr_v4i)g, (lptr_v4i)l, 0, 0);
}
#endif

// ---------------------------------------------------------------------------
// fp32 -> f16 cast (memory bound, trivial)
// ---------------------------------------------------------------------------
__global__ void cast_f32_to_f16_k(const float* __restrict__ src,
                                  _Float16* __restrict__ dst, int n) {
    int i = blockIdx.x * blockDim.x + threadIdx.x;
    if (i < n) dst[i] = (_Float16)src[i];
}

// W3 [2048,50] fp32 -> padded f16 [2048,64]
__global__ void pad_w3_k(const float* __restrict__ src, _Float16* __restrict__ dst) {
    int i = blockIdx.x * blockDim.x + threadIdx.x;
    if (i < 2048 * 64) {
        int k = i >> 6, n = i & 63;
        dst[i] = (n < 50) ? (_Float16)src[k * 50 + n] : (_Float16)0.0f;
    }
}

// ---------------------------------------------------------------------------
// Tiled f16 WMMA GEMM:  C = act(A[MxK] * B[KxN] + bias)
//   Double-buffered LDS, one barrier per K-chunk (KC=32):
//     prefetch chunk i+1 (A async-to-LDS if available, B -> regs),
//     gather + WMMA from buffer[i&1], commit prefetch to buffer[~i&1], barrier.
//   B is staged K-transposed in LDS so both fragment gathers are two aligned
//   ds_load_b128 (LDS row stride padded to 40 halves -> conflict-free).
// ---------------------------------------------------------------------------
template <int WAVES_M, int WAVES_N, int WM, int WN, bool RELU, bool OUT16>
__global__ __launch_bounds__(WAVES_M * WAVES_N * 32)
void gemm_wmma_k(const _Float16* __restrict__ A, const _Float16* __restrict__ Bm,
                 const float* __restrict__ bias, void* __restrict__ Cout,
                 int M, int N, int K, int Nstore) {
    constexpr int KC  = 32;
    constexpr int BM  = WAVES_M * WM * 16;
    constexpr int BN  = WAVES_N * WN * 16;
    constexpr int T   = WAVES_M * WAVES_N * 32;
    constexpr int LDA = KC + 8;   // 40 halves: pad to avoid bank conflicts
    constexpr int LDB = KC + 8;
    constexpr int AIT = BM * KC / (T * 8);
    constexpr int BIT = KC * BN / (T * 8);

    __shared__ _Float16 As[2][BM * LDA];       // [buf][row][k]
    __shared__ _Float16 Bs[2][BN * LDB];       // [buf][col][k] (transposed)

    const int tid  = threadIdx.x;
    const int wave = tid >> 5;
    const int lane = tid & 31;
    const int wm   = wave / WAVES_N;
    const int wn   = wave % WAVES_N;
    const int m0   = blockIdx.y * BM;
    const int n0   = blockIdx.x * BN;

    const int lh  = lane & 15;
    const int kbA = (lane < 16) ? 0 : 8;    // 16-bit A layout: K {kbA..+7, kbA+16..+23}
    const int kbB = (lane < 16) ? 0 : 8;    // 16-bit B layout: K {kbB..+7, kbB+16..+23}

    v8f acc[WM][WN];
    const v8f zero = {0.f, 0.f, 0.f, 0.f, 0.f, 0.f, 0.f, 0.f};
#pragma unroll
    for (int i = 0; i < WM; ++i)
#pragma unroll
        for (int j = 0; j < WN; ++j) acc[i][j] = zero;

    const int nch = K / KC;

    // ---- stage chunk 0 into buffer 0 ----
    {
#if USE_ASYNC_LDS
#pragma unroll
        for (int it = 0; it < AIT; ++it) {
            int e = (it * T + tid) * 8, r = e / KC, kk = e % KC;
            async_copy_b128(A + (size_t)(m0 + r) * K + kk, &As[0][r * LDA + kk]);
        }
#else
#pragma unroll
        for (int it = 0; it < AIT; ++it) {
            int e = (it * T + tid) * 8, r = e / KC, kk = e % KC;
            *(h8*)(&As[0][r * LDA + kk]) = *(const h8*)(A + (size_t)(m0 + r) * K + kk);
        }
#endif
#pragma unroll
        for (int it = 0; it < BIT; ++it) {
            int e = (it * T + tid) * 8, kk = e / BN, c = e % BN;
            h8 v = *(const h8*)(Bm + (size_t)kk * N + n0 + c);
#pragma unroll
            for (int q = 0; q < 8; ++q) Bs[0][(c + q) * LDB + kk] = v[q];
        }
#if USE_ASYNC_LDS
        __builtin_amdgcn_s_wait_asynccnt(0);
#endif
    }
    __syncthreads();

    for (int ci = 0; ci < nch; ++ci) {
        const int cur = ci & 1, nxt = cur ^ 1;
        const bool hasNext = (ci + 1 < nch);
        const int k0n = (ci + 1) * KC;

        // ---- issue next-chunk prefetch first (overlaps with gathers + WMMA) ----
        h8 bPre[BIT];
#if !USE_ASYNC_LDS
        h8 aPre[AIT];
#endif
        if (hasNext) {
#if USE_ASYNC_LDS
#pragma unroll
            for (int it = 0; it < AIT; ++it) {
                int e = (it * T + tid) * 8, r = e / KC, kk = e % KC;
                async_copy_b128(A + (size_t)(m0 + r) * K + k0n + kk,
                                &As[nxt][r * LDA + kk]);
            }
#else
#pragma unroll
            for (int it = 0; it < AIT; ++it) {
                int e = (it * T + tid) * 8, r = e / KC, kk = e % KC;
                aPre[it] = *(const h8*)(A + (size_t)(m0 + r) * K + k0n + kk);
            }
#endif
#pragma unroll
            for (int it = 0; it < BIT; ++it) {
                int e = (it * T + tid) * 8, kk = e / BN, c = e % BN;
                bPre[it] = *(const h8*)(Bm + (size_t)(k0n + kk) * N + n0 + c);
            }
        }

        // ---- gather fragments from current buffer ----
        v16h af[WM], bf[WN];
#pragma unroll
        for (int i = 0; i < WM; ++i) {
            int r = (wm * WM + i) * 16 + lh;
            h8 lo = *(const h8*)(&As[cur][r * LDA + kbA]);
            h8 hi = *(const h8*)(&As[cur][r * LDA + kbA + 16]);
#pragma unroll
            for (int q = 0; q < 8; ++q) { af[i][q] = lo[q]; af[i][8 + q] = hi[q]; }
        }
#pragma unroll
        for (int j = 0; j < WN; ++j) {
            int c = (wn * WN + j) * 16 + lh;
            h8 lo = *(const h8*)(&Bs[cur][c * LDB + kbB]);
            h8 hi = *(const h8*)(&Bs[cur][c * LDB + kbB + 16]);
#pragma unroll
            for (int q = 0; q < 8; ++q) { bf[j][q] = lo[q]; bf[j][8 + q] = hi[q]; }
        }

        // ---- 16x16x32 f16 WMMA, fp32 accumulate ----
#pragma unroll
        for (int i = 0; i < WM; ++i)
#pragma unroll
            for (int j = 0; j < WN; ++j)
                acc[i][j] = __builtin_amdgcn_wmma_f32_16x16x32_f16(
                    false, af[i], false, bf[j], (short)0, acc[i][j], false, false);

        // ---- commit prefetched data into the next buffer ----
        if (hasNext) {
#if !USE_ASYNC_LDS
#pragma unroll
            for (int it = 0; it < AIT; ++it) {
                int e = (it * T + tid) * 8, r = e / KC, kk = e % KC;
                *(h8*)(&As[nxt][r * LDA + kk]) = aPre[it];
            }
#endif
#pragma unroll
            for (int it = 0; it < BIT; ++it) {
                int e = (it * T + tid) * 8, kk = e / BN, c = e % BN;
#pragma unroll
                for (int q = 0; q < 8; ++q) Bs[nxt][(c + q) * LDB + kk] = bPre[it][q];
            }
#if USE_ASYNC_LDS
            __builtin_amdgcn_s_wait_asynccnt(0);
#endif
        }
        __syncthreads();
    }

    // ---- epilogue: bias + optional ReLU, f16 or f32 store (C/D VGPR layout) ----
    const int mbase = (lane < 16) ? 0 : 8;
#pragma unroll
    for (int i = 0; i < WM; ++i) {
#pragma unroll
        for (int j = 0; j < WN; ++j) {
            int gm = m0 + (wm * WM + i) * 16;
            int gn = n0 + (wn * WN + j) * 16 + lh;
            if (gn < Nstore) {
                float bv = bias[gn];
#pragma unroll
                for (int r = 0; r < 8; ++r) {
                    float v = acc[i][j][r] + bv;
                    if (RELU) v = fmaxf(v, 0.0f);
                    int row = gm + mbase + r;
                    if (OUT16)
                        ((_Float16*)Cout)[(size_t)row * N + gn] = (_Float16)v;
                    else
                        ((float*)Cout)[(size_t)row * N + gn] = v;
                }
            }
        }
    }
}

// ---------------------------------------------------------------------------
// Geometry tail: one wave32 per batch element.
//   lanes 0..15: solve one 8x8 homography system each (partial-pivot GE)
//   all lanes  : warp 400 points, fold via LDS float atomics,
//                boundary rescale, subtract dense grid, store [2,17,17].
// ---------------------------------------------------------------------------
__global__ __launch_bounds__(32)
void geom_k(const float* __restrict__ out3 /*[B,64] padded, 50 valid*/,
            float* __restrict__ dout /*[B,2,17,17]*/) {
    const int b = blockIdx.x;
    const int l = threadIdx.x;

    __shared__ float flow[52];      // [2*5*5]
    __shared__ float Hs[16][9];     // homography per patch
    __shared__ float fold[578];     // [2][17][17]

    for (int i = l; i < 50; i += 32) flow[i] = out3[(size_t)b * 64 + i];
    for (int i = l; i < 578; i += 32) fold[i] = 0.0f;
    __syncthreads();

    if (l < 16) {
        const int pi = l >> 2, pj = l & 3;
        float A[8][9];
#pragma unroll
        for (int q = 0; q < 4; ++q) {
            int r = pi + (q >> 1), c = pj + (q & 1);
            float sx = 128.0f * r, sy = 128.0f * c;          // sparse grid, step 512/4
            float dx = sx + flow[r * 5 + c];
            float dy = sy + flow[25 + r * 5 + c];
            float* a0 = A[q];
            float* a1 = A[4 + q];
            a0[0] = sx; a0[1] = sy; a0[2] = 1.f; a0[3] = 0.f; a0[4] = 0.f; a0[5] = 0.f;
            a0[6] = -dx * sx; a0[7] = -dx * sy; a0[8] = dx;
            a1[0] = 0.f; a1[1] = 0.f; a1[2] = 0.f; a1[3] = sx; a1[4] = sy; a1[5] = 1.f;
            a1[6] = -dy * sx; a1[7] = -dy * sy; a1[8] = dy;
        }
        // Gaussian elimination with partial pivoting
        for (int k = 0; k < 8; ++k) {
            int p = k;
            float best = fabsf(A[k][k]);
            for (int r = k + 1; r < 8; ++r) {
                float t = fabsf(A[r][k]);
                if (t > best) { best = t; p = r; }
            }
            if (p != k) {
                for (int c = k; c < 9; ++c) {
                    float t = A[k][c]; A[k][c] = A[p][c]; A[p][c] = t;
                }
            }
            float inv = 1.0f / A[k][k];
            for (int r = k + 1; r < 8; ++r) {
                float f = A[r][k] * inv;
                for (int c = k; c < 9; ++c) A[r][c] -= f * A[k][c];
            }
        }
        float h[8];
        for (int k = 7; k >= 0; --k) {
            float s = A[k][8];
            for (int c = k + 1; c < 8; ++c) s -= A[k][c] * h[c];
            h[k] = s / A[k][k];
        }
        for (int t = 0; t < 8; ++t) Hs[l][t] = h[t];
        Hs[l][8] = 1.0f;
    }
    __syncthreads();

    // warp 16 patches x 25 kernel points; fold-accumulate
    for (int idx = l; idx < 400; idx += 32) {
        int patch = idx / 25;
        int kp    = idx - patch * 25;
        int pi = patch >> 2, pj = patch & 3;
        int kh = kp / 5, kw = kp - kh * 5;
        int gi = pi * 4 + kh, gj = pj * 4 + kw;
        float px = 32.0f * gi, py = 32.0f * gj;               // dense grid, step 512/16
        const float* Hm = Hs[patch];
        float w  = Hm[6] * px + Hm[7] * py + Hm[8];
        float iw = 1.0f / w;
        float u  = (Hm[0] * px + Hm[1] * py + Hm[2]) * iw;
        float v  = (Hm[3] * px + Hm[4] * py + Hm[5]) * iw;
        atomicAdd(&fold[gi * 17 + gj], u);
        atomicAdd(&fold[289 + gi * 17 + gj], v);
    }
    __syncthreads();

    // boundary rescale, subtract grid, write out
    for (int i = l; i < 578; i += 32) {
        int ch = i / 289;
        int rc = i - ch * 289;
        int r = rc / 17, c = rc - r * 17;
        float val = fold[i];
        if ((r & 3) == 0 && (c & 3) == 0) val *= 0.5f;                       // crossings
        if ((r == 0 || r == 16) && (c == 0 || c == 16)) val *= 2.0f;         // corners
        float g = (ch == 0) ? 32.0f * r : 32.0f * c;
        dout[(size_t)b * 578 + i] = val - g;
    }
}

// ---------------------------------------------------------------------------
extern "C" void kernel_launch(void* const* d_in, const int* in_sizes, int n_in,
                              void* d_out, int out_size, void* d_ws, size_t ws_size,
                              hipStream_t stream) {
    const float* x  = (const float*)d_in[0];   // [8192,1024]
    const float* W1 = (const float*)d_in[1];   // [1024,1024]
    const float* b1 = (const float*)d_in[2];   // [1024]
    const float* W2 = (const float*)d_in[3];   // [1024,2048]
    const float* b2 = (const float*)d_in[4];   // [2048]
    const float* W3 = (const float*)d_in[5];   // [2048,50]
    const float* b3 = (const float*)d_in[6];   // [50]

    const size_t Bn = 8192, K1 = 1024, N1 = 1024, N2 = 2048, N3P = 64;

    // workspace layout (256B aligned)
    size_t off = 0;
    auto take = [&](size_t bytes) { size_t o = off; off += (bytes + 255) & ~(size_t)255; return o; };
    char* ws = (char*)d_ws;
    _Float16* xh  = (_Float16*)(ws + take(Bn * K1 * 2));
    _Float16* w1h = (_Float16*)(ws + take(K1 * N1 * 2));
    _Float16* w2h = (_Float16*)(ws + take(N1 * N2 * 2));
    _Float16* w3h = (_Float16*)(ws + take(N2 * N3P * 2));
    _Float16* h1h = (_Float16*)(ws + take(Bn * N1 * 2));
    _Float16* h2h = (_Float16*)(ws + take(Bn * N2 * 2));
    float*    o3  = (float*)   (ws + take(Bn * N3P * 4));
    (void)ws_size;

    // casts
    {
        int n = (int)(Bn * K1);
        cast_f32_to_f16_k<<<(n + 255) / 256, 256, 0, stream>>>(x, xh, n);
        n = (int)(K1 * N1);
        cast_f32_to_f16_k<<<(n + 255) / 256, 256, 0, stream>>>(W1, w1h, n);
        n = (int)(N1 * N2);
        cast_f32_to_f16_k<<<(n + 255) / 256, 256, 0, stream>>>(W2, w2h, n);
        n = (int)(N2 * N3P);
        pad_w3_k<<<(n + 255) / 256, 256, 0, stream>>>(W3, w3h);
    }

    // GEMM1: h1 = relu(x @ W1 + b1)   [8192,1024]
    gemm_wmma_k<2, 4, 4, 2, true, true>
        <<<dim3((unsigned)(N1 / 128), (unsigned)(Bn / 128)), 256, 0, stream>>>(
            xh, w1h, b1, h1h, (int)Bn, (int)N1, (int)K1, (int)N1);

    // GEMM2: h2 = relu(h1 @ W2 + b2)  [8192,2048]
    gemm_wmma_k<2, 4, 4, 2, true, true>
        <<<dim3((unsigned)(N2 / 128), (unsigned)(Bn / 128)), 256, 0, stream>>>(
            h1h, w2h, b2, h2h, (int)Bn, (int)N2, (int)N1, (int)N2);

    // GEMM3: o3 = h2 @ W3 + b3        [8192,64 padded, 50 valid], fp32 out
    gemm_wmma_k<8, 1, 2, 4, false, false>
        <<<dim3(1, (unsigned)(Bn / 256)), 256, 0, stream>>>(
            h2h, w3h, b3, o3, (int)Bn, (int)N3P, (int)N2, 50);

    // geometry tail: homographies + warp + fold + grid subtract
    geom_k<<<(unsigned)Bn, 32, 0, stream>>>(o3, (float*)d_out);
}
